// DeformableTransformerEncoderLayer_54391465837194
// MI455X (gfx1250) — matrix operand order repaired and checked by reading, not compiled, not run on hardware
//
#include <hip/hip_runtime.h>

// ---------------------------------------------------------------------------
// Problem constants (fixed by the reference)
// ---------------------------------------------------------------------------
#define BB   4
#define LQ   13294
#define MM   (BB * LQ)        // 53176 rows
#define DM   256
#define DFFN 1024
#define NL   4
#define NH   8
#define NP   4
#define HD   32               // DM / NH

typedef __attribute__((ext_vector_type(16))) _Float16 v16h;
typedef __attribute__((ext_vector_type(8)))  _Float16 v8h;
typedef __attribute__((ext_vector_type(8)))  float    v8f;

// ---------------------------------------------------------------------------
// Prep: qh = f16(src + pos), srch = f16(src)
// ---------------------------------------------------------------------------
__global__ void prep_kernel(const float* __restrict__ src,
                            const float* __restrict__ pos,
                            _Float16* __restrict__ qh,
                            _Float16* __restrict__ srch, int n) {
    int i = blockIdx.x * blockDim.x + threadIdx.x;
    if (i >= n) return;
    float s = src[i];
    qh[i]   = (_Float16)(s + pos[i]);
    srch[i] = (_Float16)s;
}

// Convert weight [K,N] f32 -> transposed [N,K] f16 (so GEMM B-fragments are
// K-contiguous per lane -> ds_load_b128 instead of scalar ds_load_u16).
__global__ void cvt_f16_t_kernel(const float* __restrict__ in,
                                 _Float16* __restrict__ out, int K, int N) {
    int i = blockIdx.x * blockDim.x + threadIdx.x;
    if (i >= K * N) return;
    int k = i / N, n = i - k * N;
    out[(size_t)n * K + k] = (_Float16)in[i];
}

// ---------------------------------------------------------------------------
// WMMA GEMM: out[M,N] = A[M,K](f16) @ W(T)[N,K](f16) + bias, optional ReLU.
// Block tile 128x128, BK=32, double-buffered LDS. 8 waves in a 4x2 grid; each
// wave owns 32x64 output = 2x4 v_wmma_f32_16x16x32_f16 tiles per K-step.
// Fragment layouts per CDNA5 ISA 7.12.2:
//   A 16x32 f16 : lane holds row lane%16; regs r -> K=(r/4)*16+(lane/16)*8+(r%4)*2+j
//   B 32x16 f16 : lane holds col lane%16; K=(r/4)*16+(lane/16)*8+(r%4)*2+j (same as A)
//   C 16x16 f32 : reg r -> row (lane/16)*8 + r, col lane%16
// ---------------------------------------------------------------------------
__global__ __launch_bounds__(256)
void gemm_f16_kernel(const _Float16* __restrict__ A,
                     const _Float16* __restrict__ WtT,   // [N,K]
                     const float*   __restrict__ bias,
                     float*         __restrict__ outF,
                     _Float16*      __restrict__ outH,
                     int Mrows, int N, int K, int relu) {
    constexpr int BM = 128, BN = 128, BK = 32;
    constexpr int LDA = 40, LDB = 40;   // padded strides (80B): no 4-way bank conflict
    __shared__ _Float16 As[2][BM * LDA];
    __shared__ _Float16 Bs[2][BN * LDB];

    const int tid  = threadIdx.x;
    const int lane = tid & 31;
    const int wave = tid >> 5;
    const int wm   = wave >> 1;        // 0..3 (rows of 32)
    const int wn   = wave & 1;         // 0..1 (cols of 64)

    const int rowBase = blockIdx.x * BM;
    const int colBase = blockIdx.y * BN;

    // Global->LDS staging maps
    const int arow = tid >> 1;                  // 0..127
    const int acol = (tid & 1) * 16;            // 0 or 16 halves
    int asrc = rowBase + arow; if (asrc >= Mrows) asrc = Mrows - 1;
    const _Float16* agp = A + (size_t)asrc * K + acol;

    const int bn = tid >> 1;                    // 0..127
    const int bk = (tid & 1) * 16;              // 0 or 16 halves
    const _Float16* bgp = WtT + (size_t)(colBase + bn) * K + bk;

    v8f acc[2][4] = {};

    // Prologue: stage tile 0 into buffer 0 (all loads first, then stores)
    {
        uint4 ar0 = *(const uint4*)(agp);
        uint4 ar1 = *(const uint4*)(agp + 8);
        uint4 br0 = *(const uint4*)(bgp);
        uint4 br1 = *(const uint4*)(bgp + 8);
        *(uint4*)(As[0] + arow * LDA + acol)     = ar0;
        *(uint4*)(As[0] + arow * LDA + acol + 8) = ar1;
        *(uint4*)(Bs[0] + bn * LDB + bk)         = br0;
        *(uint4*)(Bs[0] + bn * LDB + bk + 8)     = br1;
        __builtin_prefetch(agp + BK, 0, 1);
        __builtin_prefetch(bgp + BK, 0, 1);
    }

    const int nk = K / BK;
    int cur = 0;
    for (int kt = 0; kt < nk; ++kt) {
        __syncthreads();                 // buffer 'cur' visible to all waves

        // Issue next tile's global loads early; they overlap the WMMA work.
        const bool more = (kt + 1 < nk);
        uint4 ar0, ar1, br0, br1;
        if (more) {
            const int k0 = (kt + 1) * BK;
            ar0 = *(const uint4*)(agp + k0);
            ar1 = *(const uint4*)(agp + k0 + 8);
            br0 = *(const uint4*)(bgp + k0);
            br1 = *(const uint4*)(bgp + k0 + 8);
            if (kt + 2 < nk) {
                __builtin_prefetch(agp + k0 + BK, 0, 1);
                __builtin_prefetch(bgp + k0 + BK, 0, 1);
            }
        }

        // A fragments: two 16B chunks at k = kh and k = 16+kh
        v16h afr[2], bfr[4];
        const int mrow0 = wm * 32 + (lane & 15);
        const int kh    = (lane >> 4) * 8;
#pragma unroll
        for (int t = 0; t < 2; ++t) {
            const _Float16* ap = As[cur] + (mrow0 + t * 16) * LDA;
            v8h lo = *(const v8h*)(ap + kh);
            v8h hi = *(const v8h*)(ap + 16 + kh);
            afr[t] = __builtin_shufflevector(lo, hi, 0, 1, 2, 3, 4, 5, 6, 7,
                                             8, 9, 10, 11, 12, 13, 14, 15);
        }
        // B fragments: two 16B chunks at column n, k = kb and k = 16+kb
        const int kb = (lane >> 4) * 8;
#pragma unroll
        for (int t = 0; t < 4; ++t) {
            const _Float16* bp =
                Bs[cur] + (wn * 64 + t * 16 + (lane & 15)) * LDB + kb;
            v8h lo = *(const v8h*)(bp);
            v8h hi = *(const v8h*)(bp + 16);
            bfr[t] = __builtin_shufflevector(lo, hi, 0, 1, 2, 3, 4, 5, 6, 7,
                                             8, 9, 10, 11, 12, 13, 14, 15);
        }
#pragma unroll
        for (int i = 0; i < 2; ++i)
#pragma unroll
            for (int j = 0; j < 4; ++j)
                acc[i][j] = __builtin_amdgcn_wmma_f32_16x16x32_f16(
                    false, afr[i], false, bfr[j], (short)0, acc[i][j],
                    false, false);

        // Write next tile into the other buffer (no barrier needed here:
        // readers of 'cur' and writers of 'cur^1' don't collide; the barrier
        // at the top of the next iteration publishes these stores).
        if (more) {
            const int nxt = cur ^ 1;
            *(uint4*)(As[nxt] + arow * LDA + acol)     = ar0;
            *(uint4*)(As[nxt] + arow * LDA + acol + 8) = ar1;
            *(uint4*)(Bs[nxt] + bn * LDB + bk)         = br0;
            *(uint4*)(Bs[nxt] + bn * LDB + bk + 8)     = br1;
        }
        cur ^= 1;
    }

    // Epilogue
    const int rb = rowBase + wm * 32;
    const int nb = colBase + wn * 64;
#pragma unroll
    for (int i = 0; i < 2; ++i) {
#pragma unroll
        for (int j = 0; j < 4; ++j) {
            int col = nb + j * 16 + (lane & 15);
            float bv = bias[col];
#pragma unroll
            for (int r = 0; r < 8; ++r) {
                int row = rb + i * 16 + ((lane >> 4) << 3) + r;
                if (row < Mrows) {
                    float v = acc[i][j][r] + bv;
                    if (relu) v = v > 0.f ? v : 0.f;
                    if (outF) outF[(size_t)row * N + col] = v;
                    if (outH) outH[(size_t)row * N + col] = (_Float16)v;
                }
            }
        }
    }
}

// ---------------------------------------------------------------------------
// Softmax over the 16 (NL*NP) logits of each (row, head); in place on [M,128]
// ---------------------------------------------------------------------------
__global__ void softmax16_kernel(float* __restrict__ aw, int total) {
    int i = blockIdx.x * blockDim.x + threadIdx.x;
    if (i >= total) return;
    float* p = aw + (size_t)(i >> 3) * (NH * NL * NP) + (i & 7) * (NL * NP);
    float v[16], mx = -1e30f;
#pragma unroll
    for (int t = 0; t < 16; ++t) { v[t] = p[t]; mx = fmaxf(mx, v[t]); }
    float s = 0.f;
#pragma unroll
    for (int t = 0; t < 16; ++t) { v[t] = __expf(v[t] - mx); s += v[t]; }
    float inv = 1.f / s;
#pragma unroll
    for (int t = 0; t < 16; ++t) p[t] = v[t] * inv;
}

// ---------------------------------------------------------------------------
// MS-deformable sampling. One wave32 per (query row m, head h); lane = channel.
// valF:[M,256] offF:[M,256] aw:[M,128] ref:[B,LQ,NL,2] -> sampH f16 [M,256]
// ---------------------------------------------------------------------------
__global__ __launch_bounds__(256)
void msdeform_kernel(const float* __restrict__ valF,
                     const float* __restrict__ offF,
                     const float* __restrict__ aw,
                     const float* __restrict__ ref,
                     _Float16* __restrict__ sampH) {
    const int gw   = (blockIdx.x * blockDim.x + threadIdx.x) >> 5;
    const int lane = threadIdx.x & 31;
    if (gw >= MM * NH) return;
    const int m = gw >> 3;          // row (b*LQ + q)
    const int h = gw & 7;
    const int b = m / LQ;

    const int lvlH[NL]   = {100, 50, 25, 13};
    const int lvlW[NL]   = {100, 50, 25, 13};
    const int lvlOff[NL] = {0, 10000, 12500, 13125};

    float acc = 0.f;
#pragma unroll
    for (int l = 0; l < NL; ++l) {
        const int   W = lvlW[l], H = lvlH[l];
        const float fW = (float)W, fH = (float)H;
        const float rx = ref[((size_t)m * NL + l) * 2 + 0];
        const float ry = ref[((size_t)m * NL + l) * 2 + 1];
        const size_t vbase = (size_t)(b * LQ + lvlOff[l]) * DM + h * HD + lane;
#pragma unroll
        for (int p = 0; p < NP; ++p) {
            const int oc = ((h * NL + l) * NP + p) * 2;
            const float ox = offF[(size_t)m * DM + oc];
            const float oy = offF[(size_t)m * DM + oc + 1];
            const float a  = aw[(size_t)m * (NH * NL * NP) + (h * NL + l) * NP + p];
            const float x = (rx + ox / fW) * fW - 0.5f;
            const float y = (ry + oy / fH) * fH - 0.5f;
            const float x0f = floorf(x), y0f = floorf(y);
            const float fx = x - x0f, fy = y - y0f;
            const int x0 = (int)x0f, y0 = (int)y0f;
#pragma unroll
            for (int dy = 0; dy < 2; ++dy) {
#pragma unroll
                for (int dx = 0; dx < 2; ++dx) {
                    const int xi = x0 + dx, yi = y0 + dy;
                    if (xi >= 0 && xi < W && yi >= 0 && yi < H) {
                        const float w = (dx ? fx : 1.f - fx) * (dy ? fy : 1.f - fy);
                        const float g = valF[vbase + (size_t)(yi * W + xi) * DM];
                        acc += a * w * g;
                    }
                }
            }
        }
    }
    sampH[(size_t)m * DM + h * HD + lane] = (_Float16)acc;
}

// ---------------------------------------------------------------------------
// Fused residual add + LayerNorm over DM=256; one block per row.
// ---------------------------------------------------------------------------
__global__ __launch_bounds__(256)
void add_ln_kernel(const float* __restrict__ resid,
                   const float* __restrict__ delta,
                   const float* __restrict__ g,
                   const float* __restrict__ beta,
                   float* __restrict__ outF,
                   _Float16* __restrict__ outH) {
    __shared__ float sbuf[DM];
    const int row = blockIdx.x, c = threadIdx.x;
    const size_t base = (size_t)row * DM;
    float v = resid[base + c] + delta[base + c];
    sbuf[c] = v; __syncthreads();
    for (int s = DM / 2; s > 0; s >>= 1) {
        if (c < s) sbuf[c] += sbuf[c + s];
        __syncthreads();
    }
    const float mean = sbuf[0] * (1.0f / DM);
    __syncthreads();
    const float d = v - mean;
    sbuf[c] = d * d; __syncthreads();
    for (int s = DM / 2; s > 0; s >>= 1) {
        if (c < s) sbuf[c] += sbuf[c + s];
        __syncthreads();
    }
    const float var = sbuf[0] * (1.0f / DM);
    const float r = rsqrtf(var + 1e-5f);
    const float o = d * r * g[c] + beta[c];
    outF[base + c] = o;
    if (outH) outH[base + c] = (_Float16)o;
}

// ---------------------------------------------------------------------------
// Host launcher
// ---------------------------------------------------------------------------
extern "C" void kernel_launch(void* const* d_in, const int* in_sizes, int n_in,
                              void* d_out, int out_size, void* d_ws, size_t ws_size,
                              hipStream_t stream) {
    (void)in_sizes; (void)n_in; (void)out_size; (void)ws_size;

    const float* src = (const float*)d_in[0];
    const float* pos = (const float*)d_in[1];
    const float* ref = (const float*)d_in[2];
    // d_in[3]=spatial_shapes, d_in[4]=level_start_index: compile-time constants
    const float* Wv = (const float*)d_in[5];  const float* bv = (const float*)d_in[6];
    const float* Ws = (const float*)d_in[7];  const float* bs = (const float*)d_in[8];
    const float* Wa = (const float*)d_in[9];  const float* ba = (const float*)d_in[10];
    const float* Wo = (const float*)d_in[11]; const float* bo = (const float*)d_in[12];
    const float* W1 = (const float*)d_in[13]; const float* b1 = (const float*)d_in[14];
    const float* W2 = (const float*)d_in[15]; const float* b2 = (const float*)d_in[16];
    const float* g1 = (const float*)d_in[17]; const float* be1 = (const float*)d_in[18];
    const float* g2 = (const float*)d_in[19]; const float* be2 = (const float*)d_in[20];
    float* out = (float*)d_out;

    const size_t M = MM;
    char* ws = (char*)d_ws;
    size_t off = 0;
    auto give = [&](size_t bytes) {
        size_t r = off; off += (bytes + 255) & ~(size_t)255; return r;
    };
    // Reused regions: QH(qh -> sampH), SH(srch -> xh), VF(valF -> ffn2out),
    // OF(offF -> attnF).
    const size_t QH = give(M * DM * 2);
    const size_t SH = give(M * DM * 2);
    const size_t VF = give(M * DM * 4);
    const size_t OF = give(M * DM * 4);
    const size_t AW = give(M * (NH * NL * NP) * 4);
    const size_t XF = give(M * DM * 4);
    const size_t HH = give(M * DFFN * 2);
    const size_t WVH = give(DM * DM * 2);
    const size_t WSH = give(DM * DM * 2);
    const size_t WAH = give(DM * (NH * NL * NP) * 2);
    const size_t WOH = give(DM * DM * 2);
    const size_t W1H = give(DM * DFFN * 2);
    const size_t W2H = give(DFFN * DM * 2);

    _Float16* qh    = (_Float16*)(ws + QH);
    _Float16* sampH = (_Float16*)(ws + QH);
    _Float16* srch  = (_Float16*)(ws + SH);
    _Float16* xh    = (_Float16*)(ws + SH);
    float*    valF  = (float*)(ws + VF);
    float*    f2F   = (float*)(ws + VF);
    float*    offF  = (float*)(ws + OF);
    float*    attnF = (float*)(ws + OF);
    float*    awF   = (float*)(ws + AW);
    float*    xF    = (float*)(ws + XF);
    _Float16* hh    = (_Float16*)(ws + HH);
    _Float16* wvh = (_Float16*)(ws + WVH); _Float16* wsh = (_Float16*)(ws + WSH);
    _Float16* wah = (_Float16*)(ws + WAH); _Float16* woh = (_Float16*)(ws + WOH);
    _Float16* w1h = (_Float16*)(ws + W1H); _Float16* w2h = (_Float16*)(ws + W2H);

    // 1) prep + weight conversion (transposed to [N,K] f16)
    {
        int n = (int)(M * DM);
        prep_kernel<<<(n + 255) / 256, 256, 0, stream>>>(src, pos, qh, srch, n);
        cvt_f16_t_kernel<<<(DM*DM + 255)/256, 256, 0, stream>>>(Wv, wvh, DM, DM);
        cvt_f16_t_kernel<<<(DM*DM + 255)/256, 256, 0, stream>>>(Ws, wsh, DM, DM);
        cvt_f16_t_kernel<<<(DM*128 + 255)/256, 256, 0, stream>>>(Wa, wah, DM, 128);
        cvt_f16_t_kernel<<<(DM*DM + 255)/256, 256, 0, stream>>>(Wo, woh, DM, DM);
        cvt_f16_t_kernel<<<(DM*DFFN + 255)/256, 256, 0, stream>>>(W1, w1h, DM, DFFN);
        cvt_f16_t_kernel<<<(DFFN*DM + 255)/256, 256, 0, stream>>>(W2, w2h, DFFN, DM);
    }

    const int mblk = (int)((M + 127) / 128);
    // 2) value = srch @ Wv + bv
    gemm_f16_kernel<<<dim3(mblk, DM / 128), 256, 0, stream>>>(
        srch, wvh, bv, valF, nullptr, (int)M, DM, DM, 0);
    // 3) off = qh @ Ws + bs
    gemm_f16_kernel<<<dim3(mblk, DM / 128), 256, 0, stream>>>(
        qh, wsh, bs, offF, nullptr, (int)M, DM, DM, 0);
    // 4) aw logits = qh @ Wa + ba
    gemm_f16_kernel<<<dim3(mblk, 1), 256, 0, stream>>>(
        qh, wah, ba, awF, nullptr, (int)M, 128, DM, 0);
    // 5) softmax per (row, head)
    {
        int total = (int)(M * NH);
        softmax16_kernel<<<(total + 255) / 256, 256, 0, stream>>>(awF, total);
    }
    // 6) deformable sampling -> sampH (reuses qh region; qh consumed above)
    {
        int waves = (int)(M * NH);
        msdeform_kernel<<<(waves + 7) / 8, 256, 0, stream>>>(
            valF, offF, awF, ref, sampH);
    }
    // 7) attn = sampH @ Wo + bo   (writes into offF region; offF consumed)
    gemm_f16_kernel<<<dim3(mblk, DM / 128), 256, 0, stream>>>(
        sampH, woh, bo, attnF, nullptr, (int)M, DM, DM, 0);
    // 8) x = LN(src + attn); also f16 copy for FFN
    add_ln_kernel<<<(int)M, 256, 0, stream>>>(src, attnF, g1, be1, xF, xh);
    // 9) h = relu(x @ W1 + b1) -> f16
    gemm_f16_kernel<<<dim3(mblk, DFFN / 128), 256, 0, stream>>>(
        xh, w1h, b1, nullptr, hh, (int)M, DFFN, DM, 1);
    // 10) f2 = h @ W2 + b2   (writes into valF region; value consumed)
    gemm_f16_kernel<<<dim3(mblk, DM / 128), 256, 0, stream>>>(
        hh, w2h, b2, f2F, nullptr, (int)M, DM, DFFN, 0);
    // 11) out = LN(x + f2)
    add_ln_kernel<<<(int)M, 256, 0, stream>>>(xF, f2F, g2, be2, out, nullptr);
}
